// DeepSeekMoELayer_55783035240605
// MI455X (gfx1250) — matrix operand order repaired and checked by reading, not compiled, not run on hardware
//
#include <hip/hip_runtime.h>
#include <hip/hip_bf16.h>

// ---------------------------------------------------------------------------
// Types for CDNA5 WMMA (wave32): v_wmma_f32_16x16x32_bf16
// ---------------------------------------------------------------------------
typedef __bf16 bf16;
typedef bf16  v8bf  __attribute__((ext_vector_type(8)));
typedef bf16  v16bf __attribute__((ext_vector_type(16)));
typedef float v8f   __attribute__((ext_vector_type(8)));

#define D_MODEL   1024
#define D_HIDDEN  2048
#define N_TOKENS  4096
#define N_EXPERTS 8
#define CAPACITY  640       // ceil(4096/8 * 1.25)
#define LDS_STRIDE 40       // bf16 elems per LDS row (32 data + 8 pad), 80B, 16B-aligned

// Load a 16-element bf16 fragment from two contiguous 8-element chunks.
__device__ __forceinline__ v16bf ld16(const bf16* p0, const bf16* p1) {
  v8bf lo = *(const v8bf*)p0;
  v8bf hi = *(const v8bf*)p1;
  v16bf r;
#pragma unroll
  for (int i = 0; i < 8; ++i) { r[i] = lo[i]; r[i + 8] = hi[i]; }
  return r;
}

// CDNA5 async global->LDS copy of 16 bytes (tracked by ASYNCcnt).
__device__ __forceinline__ void async_cp16(uint32_t lds_byte_off, const bf16* gaddr) {
  asm volatile("global_load_async_to_lds_b128 %0, %1, off"
               :: "v"(lds_byte_off), "v"(gaddr) : "memory");
}
__device__ __forceinline__ void wait_async0() {
  asm volatile("s_wait_asynccnt 0" ::: "memory");
}
__device__ __forceinline__ uint32_t lds_off(const void* p) {
  // LDS aperture: flat_addr[31:0] == wave-relative LDS byte offset
  return (uint32_t)(uintptr_t)p;
}

// ---------------------------------------------------------------------------
// 1) fp32 -> bf16 staging (8 elements / thread, vectorized)
// ---------------------------------------------------------------------------
__global__ __launch_bounds__(256) void cvt_kernel(const float* __restrict__ in,
                                                  bf16* __restrict__ out, int n) {
  size_t i = ((size_t)blockIdx.x * 256 + threadIdx.x) * 8;
  if (i + 8 <= (size_t)n) {
    float4 a = *(const float4*)(in + i);
    float4 b = *(const float4*)(in + i + 4);
    out[i + 0] = (bf16)a.x; out[i + 1] = (bf16)a.y;
    out[i + 2] = (bf16)a.z; out[i + 3] = (bf16)a.w;
    out[i + 4] = (bf16)b.x; out[i + 5] = (bf16)b.y;
    out[i + 6] = (bf16)b.z; out[i + 7] = (bf16)b.w;
  }
}

// ---------------------------------------------------------------------------
// 2) Router: one wave32 per token. fp32 logits, softmax, top-2.
// ---------------------------------------------------------------------------
__global__ __launch_bounds__(256) void router_kernel(const float* __restrict__ x,
                                                     const float* __restrict__ wr,
                                                     int* __restrict__ tidx,
                                                     float* __restrict__ tgate) {
  const int token = blockIdx.x * 8 + (threadIdx.x >> 5);
  const int lane  = threadIdx.x & 31;
  if (token >= N_TOKENS) return;
  const float* xr = x + (size_t)token * D_MODEL;
  float acc[N_EXPERTS] = {};
  for (int j = lane; j < D_MODEL; j += 32) {
    const float xv = xr[j];
#pragma unroll
    for (int e = 0; e < N_EXPERTS; ++e) acc[e] += xv * wr[e * D_MODEL + j];
  }
#pragma unroll
  for (int e = 0; e < N_EXPERTS; ++e)
#pragma unroll
    for (int off = 16; off > 0; off >>= 1) acc[e] += __shfl_xor(acc[e], off, 32);

  if (lane == 0) {
    float m = acc[0];
#pragma unroll
    for (int e = 1; e < N_EXPERTS; ++e) m = fmaxf(m, acc[e]);
    float g[N_EXPERTS], s = 0.f;
#pragma unroll
    for (int e = 0; e < N_EXPERTS; ++e) { g[e] = __expf(acc[e] - m); s += g[e]; }
    const float inv = 1.0f / s;
#pragma unroll
    for (int e = 0; e < N_EXPERTS; ++e) g[e] *= inv;
    int i0 = 0;
#pragma unroll
    for (int e = 1; e < N_EXPERTS; ++e) if (g[e] > g[i0]) i0 = e;   // first max (ties)
    int i1 = (i0 == 0) ? 1 : 0;
#pragma unroll
    for (int e = 0; e < N_EXPERTS; ++e) if (e != i0 && g[e] > g[i1]) i1 = e;
    tidx[token * 2 + 0]  = i0;  tidx[token * 2 + 1]  = i1;
    tgate[token * 2 + 0] = g[i0]; tgate[token * 2 + 1] = g[i1];
  }
}

// ---------------------------------------------------------------------------
// 3) Capacity scan: lane e owns expert e's sequential counter.
// ---------------------------------------------------------------------------
__global__ void scan_kernel(const int* __restrict__ tidx, char* __restrict__ keep) {
  const int e = threadIdx.x;
  if (e >= N_EXPERTS) return;
  int cnt = 0;
  for (int n = 0; n < N_TOKENS; ++n) {
    const int i0 = tidx[n * 2], i1 = tidx[n * 2 + 1];
    char k = 0;
    if (i0 == e || i1 == e) { k = (cnt < CAPACITY) ? 1 : 0; ++cnt; }
    keep[n * N_EXPERTS + e] = k;
  }
}

// ---------------------------------------------------------------------------
// 4) Renormalize kept gates -> dense combine[N, 8]
// ---------------------------------------------------------------------------
__global__ __launch_bounds__(256) void combine_kernel(const int* __restrict__ tidx,
                                                      const float* __restrict__ tgate,
                                                      const char* __restrict__ keep,
                                                      float* __restrict__ combine) {
  const int n = blockIdx.x * 256 + threadIdx.x;
  if (n >= N_TOKENS) return;
  const int i0 = tidx[n * 2], i1 = tidx[n * 2 + 1];
  const float g0 = tgate[n * 2]     * (float)keep[n * N_EXPERTS + i0];
  const float g1 = tgate[n * 2 + 1] * (float)keep[n * N_EXPERTS + i1];
  const float s  = g0 + g1 + 1e-6f;
  float row[N_EXPERTS] = {};
  row[i0] += g0 / s;
  row[i1] += g1 / s;
#pragma unroll
  for (int e = 0; e < N_EXPERTS; ++e) combine[n * N_EXPERTS + e] = row[e];
}

// ---------------------------------------------------------------------------
// 5) hidden = silu(x @ w_gate^T): async-LDS double-buffered bf16 WMMA.
//    Block = 128 thr (4 waves), 64x64 block tile, 32x32 per wave, K-step 32.
// ---------------------------------------------------------------------------
__global__ __launch_bounds__(128) void gate_silu_kernel(const bf16* __restrict__ xb,
                                                        const bf16* __restrict__ wgb,
                                                        bf16* __restrict__ hidden) {
  __shared__ __align__(16) bf16 smA[2][64 * LDS_STRIDE];
  __shared__ __align__(16) bf16 smB[2][64 * LDS_STRIDE];
  const int tid  = threadIdx.x;
  const int lane = tid & 31;
  const int wave = tid >> 5;
  const int half = lane >> 4, mr = lane & 15;
  const int m0 = blockIdx.y * 64;
  const int n0 = blockIdx.x * 64;
  const int wm = (wave >> 1) * 32, wn = (wave & 1) * 32;

  // Each thread fills 2 16B chunks of A and 2 of B per K-tile (256 chunks each).
  auto issue = [&](int buf, int k0) {
#pragma unroll
    for (int q = 0; q < 2; ++q) {
      const int c = tid * 2 + q, row = c >> 2, part = c & 3;
      async_cp16(lds_off(&smA[buf][row * LDS_STRIDE + part * 8]),
                 xb + (size_t)(m0 + row) * D_MODEL + k0 + part * 8);
      async_cp16(lds_off(&smB[buf][row * LDS_STRIDE + part * 8]),
                 wgb + (size_t)(n0 + row) * D_MODEL + k0 + part * 8);
    }
  };

  v8f acc[2][2] = {};
  issue(0, 0);
  for (int kt = 0; kt < D_MODEL / 32; ++kt) {
    const int buf = kt & 1;
    wait_async0();        // my fills for `buf` have landed in LDS
    __syncthreads();      // everyone's fills landed; everyone done reading buf^1
    if (kt + 1 < D_MODEL / 32) issue(buf ^ 1, (kt + 1) * 32);

    v16bf af[2], bfr[2];
#pragma unroll
    for (int i = 0; i < 2; ++i) {
      const bf16* ap = &smA[buf][(wm + i * 16 + mr) * LDS_STRIDE];
      af[i]  = ld16(ap + half * 8, ap + 16 + half * 8);
      const bf16* bp = &smB[buf][(wn + i * 16 + mr) * LDS_STRIDE];
      bfr[i] = ld16(bp + half * 8, bp + 16 + half * 8);
    }
#pragma unroll
    for (int i = 0; i < 2; ++i)
#pragma unroll
      for (int j = 0; j < 2; ++j)
        acc[i][j] = __builtin_amdgcn_wmma_f32_16x16x32_bf16(
            false, af[i], false, bfr[j], (short)0, acc[i][j], false, false);
  }
#pragma unroll
  for (int i = 0; i < 2; ++i)
#pragma unroll
    for (int j = 0; j < 2; ++j) {
      const int col = n0 + wn + j * 16 + mr;
#pragma unroll
      for (int r = 0; r < 8; ++r) {
        const int row = m0 + wm + i * 16 + half * 8 + r;
        const float h = acc[i][j][r];
        const float sv = h / (1.0f + __expf(-h));   // silu
        hidden[(size_t)row * D_HIDDEN + col] = (bf16)sv;
      }
    }
}

// ---------------------------------------------------------------------------
// 6) y = sum_e combine[:,e] * (hidden @ w_down[e]^T), fp32 out.
//    Async-LDS double buffering; block-uniform expert skip via __syncthreads_or.
// ---------------------------------------------------------------------------
__global__ __launch_bounds__(128) void down_kernel(const bf16* __restrict__ hidden,
                                                   const bf16* __restrict__ wdb,
                                                   const float* __restrict__ combine,
                                                   float* __restrict__ y) {
  __shared__ __align__(16) bf16 smA[2][64 * LDS_STRIDE];
  __shared__ __align__(16) bf16 smB[2][64 * LDS_STRIDE];
  const int tid  = threadIdx.x;
  const int lane = tid & 31;
  const int wave = tid >> 5;
  const int half = lane >> 4, mr = lane & 15;
  const int m0 = blockIdx.y * 64;
  const int n0 = blockIdx.x * 64;
  const int wm = (wave >> 1) * 32, wn = (wave & 1) * 32;

  v8f accY[2][2] = {};
  for (int e = 0; e < N_EXPERTS; ++e) {
    float sc[2][8];
    bool nz = false;
#pragma unroll
    for (int i = 0; i < 2; ++i)
#pragma unroll
      for (int r = 0; r < 8; ++r) {
        const float v =
            combine[(size_t)(m0 + wm + i * 16 + half * 8 + r) * N_EXPERTS + e];
        sc[i][r] = v;
        nz |= (v != 0.0f);
      }
    // Block-uniform skip (also serves as the inter-expert LDS reuse barrier).
    if (!__syncthreads_or(nz ? 1 : 0)) continue;

    const bf16* we = wdb + (size_t)e * D_MODEL * D_HIDDEN;
    auto issue = [&](int buf, int k0) {
#pragma unroll
      for (int q = 0; q < 2; ++q) {
        const int c = tid * 2 + q, row = c >> 2, part = c & 3;
        async_cp16(lds_off(&smA[buf][row * LDS_STRIDE + part * 8]),
                   hidden + (size_t)(m0 + row) * D_HIDDEN + k0 + part * 8);
        async_cp16(lds_off(&smB[buf][row * LDS_STRIDE + part * 8]),
                   we + (size_t)(n0 + row) * D_HIDDEN + k0 + part * 8);
      }
    };

    v8f acc[2][2] = {};
    issue(0, 0);
    for (int kt = 0; kt < D_HIDDEN / 32; ++kt) {
      const int buf = kt & 1;
      wait_async0();
      __syncthreads();
      if (kt + 1 < D_HIDDEN / 32) issue(buf ^ 1, (kt + 1) * 32);

      v16bf af[2], bfr[2];
#pragma unroll
      for (int i = 0; i < 2; ++i) {
        const bf16* ap = &smA[buf][(wm + i * 16 + mr) * LDS_STRIDE];
        af[i]  = ld16(ap + half * 8, ap + 16 + half * 8);
        const bf16* bp = &smB[buf][(wn + i * 16 + mr) * LDS_STRIDE];
        bfr[i] = ld16(bp + half * 8, bp + 16 + half * 8);
      }
#pragma unroll
      for (int i = 0; i < 2; ++i)
#pragma unroll
        for (int j = 0; j < 2; ++j)
          acc[i][j] = __builtin_amdgcn_wmma_f32_16x16x32_bf16(
              false, af[i], false, bfr[j], (short)0, acc[i][j], false, false);
    }
#pragma unroll
    for (int i = 0; i < 2; ++i)
#pragma unroll
      for (int j = 0; j < 2; ++j)
#pragma unroll
        for (int r = 0; r < 8; ++r) accY[i][j][r] += sc[i][r] * acc[i][j][r];
  }
#pragma unroll
  for (int i = 0; i < 2; ++i)
#pragma unroll
    for (int j = 0; j < 2; ++j) {
      const int col = n0 + wn + j * 16 + mr;
#pragma unroll
      for (int r = 0; r < 8; ++r) {
        const int row = m0 + wm + i * 16 + half * 8 + r;
        y[(size_t)row * D_MODEL + col] = accY[i][j][r];
      }
    }
}

// ---------------------------------------------------------------------------
// Launch
// ---------------------------------------------------------------------------
extern "C" void kernel_launch(void* const* d_in, const int* in_sizes, int n_in,
                              void* d_out, int out_size, void* d_ws, size_t ws_size,
                              hipStream_t stream) {
  const float* x        = (const float*)d_in[0];   // [2,2048,1024]
  const float* w_router = (const float*)d_in[1];   // [8,1024]
  const float* w_gate   = (const float*)d_in[2];   // [2048,1024]
  const float* w_down   = (const float*)d_in[3];   // [8,1024,2048]
  float* y = (float*)d_out;                        // [2,2048,1024] fp32

  char* ws = (char*)d_ws;
  bf16*  xb      = (bf16*)(ws + 0);            // 4096*1024*2  =  8,388,608
  bf16*  wgb     = (bf16*)(ws + 8388608);      // 2048*1024*2  =  4,194,304
  bf16*  wdb     = (bf16*)(ws + 12582912);     // 8*1024*2048*2= 33,554,432
  bf16*  hidden  = (bf16*)(ws + 46137344);     // 4096*2048*2  = 16,777,216
  int*   tidx    = (int*)(ws + 62914560);      // 4096*2*4
  float* tgate   = (float*)(ws + 62947328);    // 4096*2*4
  char*  keep    = (char*)(ws + 62980096);     // 4096*8
  float* combine = (float*)(ws + 63012864);    // 4096*8*4

  cvt_kernel<<<(N_TOKENS * D_MODEL) / 2048, 256, 0, stream>>>(x, xb, N_TOKENS * D_MODEL);
  cvt_kernel<<<(D_HIDDEN * D_MODEL) / 2048, 256, 0, stream>>>(w_gate, wgb, D_HIDDEN * D_MODEL);
  cvt_kernel<<<(N_EXPERTS * D_MODEL * D_HIDDEN) / 2048, 256, 0, stream>>>(
      w_down, wdb, N_EXPERTS * D_MODEL * D_HIDDEN);

  router_kernel<<<N_TOKENS / 8, 256, 0, stream>>>(x, w_router, tidx, tgate);
  scan_kernel<<<1, 32, 0, stream>>>(tidx, keep);
  combine_kernel<<<N_TOKENS / 256, 256, 0, stream>>>(tidx, tgate, keep, combine);

  dim3 gg(D_HIDDEN / 64, N_TOKENS / 64);
  gate_silu_kernel<<<gg, 128, 0, stream>>>(xb, wgb, hidden);

  dim3 gd(D_MODEL / 64, N_TOKENS / 64);
  down_kernel<<<gd, 128, 0, stream>>>(hidden, wdb, combine, y);
}
